// SoftClusterAssignment_49924699848825
// MI455X (gfx1250) — matrix operand order repaired and checked by reading, not compiled, not run on hardware
//
#include <hip/hip_runtime.h>
#include <cstdint>

// ---------------------------------------------------------------------------
// SoftClusterAssignment on gfx1250:
//   cross = z @ c^T via v_wmma_f32_16x16x32_bf16 (f32 accumulate)
//   diff  = ||z||^2 - 2*cross + ||c||^2 ; num = 1/(1+diff)   (ALPHA=1 -> pow==1)
//   q     = num / rowsum(num)   -- fused in-LDS normalization
// ---------------------------------------------------------------------------

typedef __attribute__((ext_vector_type(16))) __bf16 v16bf;
typedef __attribute__((ext_vector_type(8)))  __bf16 v8bf;
typedef __attribute__((ext_vector_type(8)))  float  v8f;

#define D_DIM 512
#define K_DIM 1024
#define M_BLK 32
#define Z_STRIDE 520     // bf16 elems per LDS z row (padded: 1040B = 4 banks shift/row)
#define NUM_STRIDE 1028  // f32 elems per LDS numerator row (padded vs 1024)

static constexpr int kLdsNumBytes = M_BLK * NUM_STRIDE * 4;   // 131584
static constexpr int kLdsZBytes   = M_BLK * Z_STRIDE * 2;     // 33280
static constexpr int kLdsBytes    = kLdsNumBytes + kLdsZBytes + M_BLK * 4 * 2; // ~161.3KB

__device__ __forceinline__ unsigned short f32_bf16(float f) {
    unsigned u = __float_as_uint(f);
    u += 0x7FFFu + ((u >> 16) & 1u);          // round to nearest even
    return (unsigned short)(u >> 16);
}

// ---- Kernel A: centroid f32 -> bf16 (workspace) + csq[k] = ||c_k||^2 ----
__global__ void __launch_bounds__(128)
centroid_prep_kernel(const float* __restrict__ centroid,
                     unsigned short* __restrict__ cb,
                     float* __restrict__ csq) {
    __shared__ float sred[4];
    const int row = blockIdx.x;      // 0..K-1
    const int t   = threadIdx.x;     // 0..127, 4 floats each
    const float4 v = *(const float4*)(centroid + (size_t)row * D_DIM + 4 * t);
    uint2 p;
    p.x = (unsigned)f32_bf16(v.x) | ((unsigned)f32_bf16(v.y) << 16);
    p.y = (unsigned)f32_bf16(v.z) | ((unsigned)f32_bf16(v.w) << 16);
    *(uint2*)(cb + (size_t)row * D_DIM + 4 * t) = p;
    float s = v.x * v.x + v.y * v.y + v.z * v.z + v.w * v.w;
    #pragma unroll
    for (int off = 16; off; off >>= 1) s += __shfl_xor(s, off, 32);
    if ((t & 31) == 0) sred[t >> 5] = s;
    __syncthreads();
    if (t == 0) csq[row] = sred[0] + sred[1] + sred[2] + sred[3];
}

// ---- Kernel B: fused GEMM + 1/(1+diff) + row-normalize ----
__global__ void __launch_bounds__(256)
soft_assign_kernel(const float* __restrict__ z,
                   const unsigned short* __restrict__ cb,
                   const float* __restrict__ csq,
                   float* __restrict__ out) {
    extern __shared__ char smem[];
    float*          numBuf = (float*)smem;                                 // [M_BLK][NUM_STRIDE]
    unsigned short* zbf    = (unsigned short*)(smem + kLdsNumBytes);       // [M_BLK][Z_STRIDE]
    float*          zsq    = (float*)(smem + kLdsNumBytes + kLdsZBytes);   // [M_BLK]
    float*          rowsum = zsq + M_BLK;                                  // [M_BLK]

    const int tid  = threadIdx.x;
    const int lane = tid & 31;
    const int wave = tid >> 5;
    const int row0 = blockIdx.x * M_BLK;

    if (tid < M_BLK) { zsq[tid] = 0.f; rowsum[tid] = 0.f; }
    __syncthreads();

    // ---- Stage z tile: f32 global -> bf16 LDS, accumulate ||z||^2 per row ----
    {
        const int r  = tid >> 3;            // 0..31 : one row, 8 threads/row
        const int c0 = (tid & 7) * 64;      // 64 floats per thread
        const float*    zp = z + (size_t)(row0 + r) * D_DIM + c0;
        unsigned short* lp = zbf + r * Z_STRIDE + c0;
        float s = 0.f;
        #pragma unroll
        for (int i = 0; i < 16; ++i) {
            float4 v = *(const float4*)(zp + 4 * i);
            s += v.x * v.x + v.y * v.y + v.z * v.z + v.w * v.w;
            uint2 p;
            p.x = (unsigned)f32_bf16(v.x) | ((unsigned)f32_bf16(v.y) << 16);
            p.y = (unsigned)f32_bf16(v.z) | ((unsigned)f32_bf16(v.w) << 16);
            *(uint2*)(lp + 4 * i) = p;
        }
        atomicAdd(&zsq[r], s);
    }
    __syncthreads();

    // ---- WMMA sweep over all K columns (8 waves: 2 in M x 4 in N) ----
    const int waveM = wave >> 2;            // 0..1
    const int waveN = wave & 3;             // 0..3
    const int mBase = waveM * 16;
    const int hi    = lane >> 4;            // lane half
    const int lo    = lane & 15;
    const int hi8   = hi * 8;

    float zs[8];
    #pragma unroll
    for (int r = 0; r < 8; ++r) zs[r] = zsq[mBase + r + hi8];   // C-layout M per reg

    // A-matrix (16x32 bf16) per ISA layout: lane lo -> row M; lane half selects
    // K subsets {0-7,16-23} vs {8-15,24-31}.
    const unsigned short* aRow = zbf + (mBase + lo) * Z_STRIDE + hi8;

    for (int chunk = 0; chunk < K_DIM / 64; ++chunk) {
        const int nCol = chunk * 64 + waveN * 16 + lo;           // global centroid index
        const float cs = csq[nCol];
        // B-matrix (32x16 bf16): lane lo -> column N; lane half -> K {0-7,16-23} vs {8-15,24-31}
        const unsigned short* bRow = cb + (size_t)nCol * D_DIM + hi8;

        v8f acc = {0.f, 0.f, 0.f, 0.f, 0.f, 0.f, 0.f, 0.f};
        #pragma unroll
        for (int dBase = 0; dBase < D_DIM; dBase += 32) {
            union { v16bf v; v8bf h[2]; } a;
            a.h[0] = *(const v8bf*)(aRow + dBase);        // K = d+hi8+0..7
            a.h[1] = *(const v8bf*)(aRow + dBase + 16);   // K = d+16+hi8+0..7
            union { v16bf v; v8bf h[2]; } b;
            b.h[0] = *(const v8bf*)(bRow + dBase);        // K = d+hi8+0..7
            b.h[1] = *(const v8bf*)(bRow + dBase + 16);   // K = d+16+hi8+0..7
            acc = __builtin_amdgcn_wmma_f32_16x16x32_bf16(
                      false, a.v, false, b.v, (short)0, acc, false, false);
        }

        #pragma unroll
        for (int r = 0; r < 8; ++r) {
            const int   m    = mBase + r + hi8;
            const float diff = zs[r] - 2.f * acc[r] + cs;
            const float num  = 1.f / (1.f + diff);        // ALPHA=1 => pow is identity
            numBuf[m * NUM_STRIDE + nCol] = num;
            float s = num;                                 // 16 lanes share this m
            #pragma unroll
            for (int off = 8; off; off >>= 1) s += __shfl_xor(s, off, 16);
            if (lo == 0) atomicAdd(&rowsum[m], s);
        }
    }
    __syncthreads();

    // ---- Normalize from LDS, coalesced float4 stores ----
    for (int idx = tid; idx < M_BLK * (K_DIM / 4); idx += 256) {
        const int r  = idx >> 8;            // 256 float4 per row
        const int c4 = idx & 255;
        const float inv = 1.f / rowsum[r];
        float4 v = *(const float4*)(numBuf + r * NUM_STRIDE + c4 * 4);
        v.x *= inv; v.y *= inv; v.z *= inv; v.w *= inv;
        *(float4*)(out + (size_t)(row0 + r) * K_DIM + c4 * 4) = v;
    }
}

extern "C" void kernel_launch(void* const* d_in, const int* in_sizes, int n_in,
                              void* d_out, int out_size, void* d_ws, size_t ws_size,
                              hipStream_t stream) {
    const float* z        = (const float*)d_in[0];   // [N, D] f32
    const float* centroid = (const float*)d_in[1];   // [K, D] f32
    float*       out      = (float*)d_out;           // [N, K] f32
    const int N = in_sizes[0] / D_DIM;               // 131072

    // Workspace: bf16 centroid (K*D*2 = 1 MB) then csq (K*4)
    unsigned short* cb  = (unsigned short*)d_ws;
    float*          csq = (float*)((char*)d_ws + (size_t)K_DIM * D_DIM * 2);

    centroid_prep_kernel<<<K_DIM, 128, 0, stream>>>(centroid, cb, csq);
    soft_assign_kernel<<<N / M_BLK, 256, kLdsBytes, stream>>>(z, cb, csq, out);
}
